// GreedyInference_72980084293740
// MI455X (gfx1250) — matrix operand order repaired and checked by reading, not compiled, not run on hardware
//
#include <hip/hip_runtime.h>

// ---------------- problem constants ----------------
#define B     128
#define T     512
#define H     640
#define V     4097
#define VP    4128          // vocab padded to multiple of 32 (2-wide N blocking)
#define KG    1280          // K for [emb|h] and [f_t|h1] GEMMs
#define NG    2560          // 4*H (LSTM gates)
#define NBLK  80            // persistent blocks (must be co-resident)
#define NTHR  256           // 8 waves of 32

typedef __attribute__((ext_vector_type(16))) __bf16 v16bf;
typedef __attribute__((ext_vector_type(8)))  float  v8f;

union FragBF { v16bf v; uint4 q[2]; };

struct WS {
    unsigned short *Wg;     // [NG x KG]  bf16   (W_ih | W_hh)
    unsigned short *Wj;     // [H  x KG]  bf16   (W_enc | W_pred)
    unsigned short *Wo;     // [VP x H ]  bf16   (W_out, zero padded)
    unsigned short *Emb;    // [V  x H ]  bf16
    float *bg;              // [NG]  b_ih + b_hh
    float *bj;              // [H]   b_enc + b_pred
    float *bo;              // [V]
    float *h, *c, *h1, *c1; // [B x H] each
    float *gates;           // [B x NG]
    unsigned short *A1;     // [B x KG]  bf16  ([emb|h])
    unsigned short *A2;     // [B x KG]  bf16  ([f_t|h1])
    unsigned short *Jb;     // [B x H ]  bf16  relu(joint)
    unsigned long long *amax; // [B] packed (mappedVal<<32)|(0xFFFF-n)
    int *label;             // [B]
    unsigned *bar;          // [2] barrier counter + generation
};

__device__ __forceinline__ unsigned short f2bf(float x) {
    unsigned u = __float_as_uint(x);
    u += 0x7FFFu + ((u >> 16) & 1u);        // round to nearest even
    return (unsigned short)(u >> 16);
}
__device__ __forceinline__ float sigmoidf(float x) { return 1.f / (1.f + __expf(-x)); }

// ---------------- grid-wide barrier (sense via generation counter) ----------
__device__ __forceinline__ void grid_sync(unsigned *cnt, unsigned *gen) {
    __syncthreads();
    if (threadIdx.x == 0) {
        __threadfence();
        unsigned g0 = __hip_atomic_load(gen, __ATOMIC_RELAXED, __HIP_MEMORY_SCOPE_AGENT);
        unsigned arrived = __hip_atomic_fetch_add(cnt, 1u, __ATOMIC_ACQ_REL, __HIP_MEMORY_SCOPE_AGENT);
        if (arrived == (unsigned)(NBLK - 1)) {
            __hip_atomic_store(cnt, 0u, __ATOMIC_RELAXED, __HIP_MEMORY_SCOPE_AGENT);
            __hip_atomic_fetch_add(gen, 1u, __ATOMIC_ACQ_REL, __HIP_MEMORY_SCOPE_AGENT);
        } else {
            while (__hip_atomic_load(gen, __ATOMIC_ACQUIRE, __HIP_MEMORY_SCOPE_AGENT) == g0)
                __builtin_amdgcn_s_sleep(2);
        }
    }
    __syncthreads();
}

// ------- one 16x32 output group (2 N-tiles share one A fragment) ------------
// A: MxK row-major bf16 (lda), Bm: NxK row-major bf16 (ldb)  => D = A * Bm^T
__device__ __forceinline__ void gemm_tile2(const unsigned short *A,
                                           const unsigned short *Bm,
                                           int lda, int ldb, int K,
                                           int m0, int n0, int lane,
                                           v8f &acc0, v8f &acc1) {
    const int lmod  = lane & 15;
    const int koffA = (lane < 16) ? 0 : 8;   // A: {0..7,16..23} / {8..15,24..31}
    const int koffB = (lane < 16) ? 0 : 8;   // B: {0..7,16..23} / {8..15,24..31}
    const unsigned short *ar  = A  + (size_t)(m0 + lmod) * lda + koffA;
    const unsigned short *br0 = Bm + (size_t)(n0 + lmod) * ldb + koffB;
    const unsigned short *br1 = Bm + (size_t)(n0 + 16 + lmod) * ldb + koffB;
    acc0 = {}; acc1 = {};
    for (int k0 = 0; k0 < K; k0 += 32) {
        FragBF fa, fb0, fb1;
        fa.q[0]  = *(const uint4 *)(ar + k0);
        fa.q[1]  = *(const uint4 *)(ar + k0 + 16);
        fb0.q[0] = *(const uint4 *)(br0 + k0);
        fb0.q[1] = *(const uint4 *)(br0 + k0 + 16);
        fb1.q[0] = *(const uint4 *)(br1 + k0);
        fb1.q[1] = *(const uint4 *)(br1 + k0 + 16);
        __builtin_prefetch(br0 + k0 + 128, 0, 1);   // global_prefetch_b8 ahead in K
        acc0 = __builtin_amdgcn_wmma_f32_16x16x32_bf16(
                 false, fa.v, false, fb0.v, (short)0, acc0, false, false);
        acc1 = __builtin_amdgcn_wmma_f32_16x16x32_bf16(
                 false, fa.v, false, fb1.v, (short)0, acc1, false, false);
    }
}

// ---------------- one-time prep: bf16 weights, fused biases, init state -----
__global__ void prep_kernel(const float *emb, const float *Wih, const float *Whh,
                            const float *bih, const float *bhh,
                            const float *Wenc, const float *benc,
                            const float *Wpred, const float *bpred,
                            const float *Wout, const float *bout,
                            const float *h0, const float *c0, WS ws) {
    int gt = blockIdx.x * blockDim.x + threadIdx.x;
    int NT = gridDim.x * blockDim.x;
    for (int i = gt; i < NG * KG; i += NT) {
        int n = i / KG, k = i % KG;
        float v = (k < H) ? Wih[n * H + k] : Whh[n * H + (k - H)];
        ws.Wg[i] = f2bf(v);
    }
    for (int i = gt; i < H * KG; i += NT) {
        int n = i / KG, k = i % KG;
        float v = (k < H) ? Wenc[n * H + k] : Wpred[n * H + (k - H)];
        ws.Wj[i] = f2bf(v);
    }
    for (int i = gt; i < VP * H; i += NT) {
        int n = i / H, k = i % H;
        ws.Wo[i] = f2bf((n < V) ? Wout[n * H + k] : 0.f);
    }
    for (int i = gt; i < V * H; i += NT) ws.Emb[i] = f2bf(emb[i]);
    for (int i = gt; i < NG; i += NT) ws.bg[i] = bih[i] + bhh[i];
    for (int i = gt; i < H;  i += NT) ws.bj[i] = benc[i] + bpred[i];
    for (int i = gt; i < V;  i += NT) ws.bo[i] = bout[i];
    for (int i = gt; i < B * H; i += NT) { ws.h[i] = h0[i]; ws.c[i] = c0[i]; }
    for (int i = gt; i < B; i += NT) { ws.label[i] = 0; ws.amax[i] = 0ull; }
    if (gt < 2) ws.bar[gt] = 0u;
}

// ---------------- persistent fused greedy decoder ---------------------------
__global__ __launch_bounds__(NTHR, 1)
void decode_kernel(const float *__restrict__ enc, const int *__restrict__ lens,
                   float *__restrict__ out, WS ws) {
    const int tid    = threadIdx.x;
    const int gt     = blockIdx.x * NTHR + tid;
    const int NT     = NBLK * NTHR;
    const int lane   = tid & 31;
    const int wave   = tid >> 5;
    const int gwave  = blockIdx.x * (NTHR / 32) + wave;
    const int nwaves = NBLK * (NTHR / 32);
    const int lmod   = lane & 15;
    const int lhalf  = (lane < 16) ? 0 : 8;
    unsigned *cnt = ws.bar, *gen = ws.bar + 1;

    for (int t = 0; t < T; ++t) {
        // --- P1: A1 = [emb(label) | h] in bf16; reset argmax slots ---------
        for (int i = gt; i < B * KG; i += NT) {
            int b = i / KG, k = i % KG;
            unsigned short v;
            if (k < H) v = (t == 0) ? (unsigned short)0
                                    : ws.Emb[(size_t)ws.label[b] * H + k];
            else       v = f2bf(ws.h[b * H + (k - H)]);
            ws.A1[i] = v;
        }
        for (int i = gt; i < B; i += NT) ws.amax[i] = 0ull;
        grid_sync(cnt, gen);

        // --- P2: gates = A1 @ [Wih|Whh]^T + b  (128 x 2560, K=1280) --------
        // 8 M-tiles x 80 N-groups = 640 groups = exactly 1 per wave
        for (int g = gwave; g < 8 * (NG / 32); g += nwaves) {
            int m0 = (g & 7) * 16, n0 = (g >> 3) * 32;
            v8f a0, a1;
            gemm_tile2(ws.A1, ws.Wg, KG, KG, KG, m0, n0, lane, a0, a1);
            int na = n0 + lmod, nb = n0 + 16 + lmod;
            float bga = ws.bg[na], bgb = ws.bg[nb];
#pragma unroll
            for (int r = 0; r < 8; ++r) {
                int m = m0 + r + lhalf;
                ws.gates[m * NG + na] = a0[r] + bga;
                ws.gates[m * NG + nb] = a1[r] + bgb;
            }
        }
        grid_sync(cnt, gen);

        // --- P3: LSTM pointwise -> h1,c1 ; A2 = [f_t | h1] bf16 ------------
        for (int i = gt; i < B * H; i += NT) {
            int b = i / H, j = i % H;
            const float *g = ws.gates + (size_t)b * NG;
            float gi = g[j], gf = g[H + j], gg = g[2 * H + j], go = g[3 * H + j];
            float cc = sigmoidf(gf) * ws.c[i] + sigmoidf(gi) * tanhf(gg);
            float hh = sigmoidf(go) * tanhf(cc);
            ws.c1[i] = cc; ws.h1[i] = hh;
            ws.A2[b * KG + H + j] = f2bf(hh);
            // f_t gather: encoder_output is (B,H,T)
            ws.A2[b * KG + j] = f2bf(enc[(size_t)b * H * T + (size_t)j * T + t]);
        }
        grid_sync(cnt, gen);

        // --- P4: joint = relu(A2 @ [Wenc|Wpred]^T + b) -> bf16 -------------
        for (int g = gwave; g < 8 * (H / 32); g += nwaves) {
            int m0 = (g & 7) * 16, n0 = (g >> 3) * 32;
            v8f a0, a1;
            gemm_tile2(ws.A2, ws.Wj, KG, KG, KG, m0, n0, lane, a0, a1);
            int na = n0 + lmod, nb = n0 + 16 + lmod;
            float bja = ws.bj[na], bjb = ws.bj[nb];
#pragma unroll
            for (int r = 0; r < 8; ++r) {
                int m = m0 + r + lhalf;
                float va = a0[r] + bja, vb = a1[r] + bjb;
                ws.Jb[m * H + na] = f2bf(va > 0.f ? va : 0.f);
                ws.Jb[m * H + nb] = f2bf(vb > 0.f ? vb : 0.f);
            }
        }
        grid_sync(cnt, gen);

        // --- P5: logits = Jb @ Wout^T + b_out ; fused argmax ---------------
        for (int g = gwave; g < 8 * (VP / 32); g += nwaves) {
            int m0 = (g & 7) * 16, n0 = (g >> 3) * 32;
            v8f a0, a1;
            gemm_tile2(ws.Jb, ws.Wo, H, H, H, m0, n0, lane, a0, a1);
#pragma unroll
            for (int half = 0; half < 2; ++half) {
                const v8f &acc = half ? a1 : a0;
                int n = n0 + half * 16 + lmod;
                bool valid = (n < V);
                float bo = valid ? ws.bo[n] : 0.f;
#pragma unroll
                for (int r = 0; r < 8; ++r) {
                    unsigned long long p = 0ull;
                    if (valid) {
                        unsigned u = __float_as_uint(acc[r] + bo);
                        u = (u & 0x80000000u) ? ~u : (u | 0x80000000u); // order map
                        p = ((unsigned long long)u << 32) | (unsigned)(0xFFFFu - n);
                    }
#pragma unroll
                    for (int s = 1; s < 16; s <<= 1) {  // reduce 16-lane half
                        unsigned lo = __shfl_xor((unsigned)p, s, 32);
                        unsigned hi = __shfl_xor((unsigned)(p >> 32), s, 32);
                        unsigned long long o = ((unsigned long long)hi << 32) | lo;
                        p = (o > p) ? o : p;
                    }
                    if (lmod == 0 && p)
                        atomicMax(&ws.amax[m0 + r + lhalf], p);
                }
            }
        }
        grid_sync(cnt, gen);

        // --- P6: commit h/c/label where not blank; emit token --------------
        for (int i = gt; i < B * H; i += NT) {
            int b = i / H, j = i % H;
            unsigned long long p = ws.amax[b];
            int k = 0xFFFF - (int)(p & 0xFFFFFFFFull);
            bool blank = (t >= lens[b]) || (k == 0);
            if (!blank) { ws.h[i] = ws.h1[i]; ws.c[i] = ws.c1[i]; }
            if (j == 0) {
                if (!blank) ws.label[b] = k;
                out[(size_t)b * T + t] = (float)(blank ? 0 : k);
            }
        }
        grid_sync(cnt, gen);
    }

    // final hidden / cell state outputs (shape (1,B,H) each)
    for (int i = gt; i < B * H; i += NT) {
        out[B * T + i]         = ws.h[i];
        out[B * T + B * H + i] = ws.c[i];
    }
}

// ---------------- host side --------------------------------------------------
extern "C" void kernel_launch(void *const *d_in, const int *in_sizes, int n_in,
                              void *d_out, int out_size, void *d_ws, size_t ws_size,
                              hipStream_t stream) {
    const float *enc   = (const float *)d_in[0];
    const int   *lens  = (const int *)  d_in[1];
    const float *emb   = (const float *)d_in[2];
    const float *Wih   = (const float *)d_in[3];
    const float *Whh   = (const float *)d_in[4];
    const float *bih   = (const float *)d_in[5];
    const float *bhh   = (const float *)d_in[6];
    const float *Wenc  = (const float *)d_in[7];
    const float *benc  = (const float *)d_in[8];
    const float *Wpred = (const float *)d_in[9];
    const float *bpred = (const float *)d_in[10];
    const float *Wout  = (const float *)d_in[11];
    const float *bout  = (const float *)d_in[12];
    const float *h0    = (const float *)d_in[13];
    const float *c0    = (const float *)d_in[14];
    float *out = (float *)d_out;

    char *base = (char *)d_ws;
    size_t off = 0;
    auto take = [&](size_t bytes) -> void * {
        off = (off + 255) & ~(size_t)255;
        void *r = base + off;
        off += bytes;
        return r;
    };

    WS ws;
    ws.Wg    = (unsigned short *)take((size_t)NG * KG * 2);
    ws.Wj    = (unsigned short *)take((size_t)H * KG * 2);
    ws.Wo    = (unsigned short *)take((size_t)VP * H * 2);
    ws.Emb   = (unsigned short *)take((size_t)V * H * 2);
    ws.bg    = (float *)take((size_t)NG * 4);
    ws.bj    = (float *)take((size_t)H * 4);
    ws.bo    = (float *)take((size_t)V * 4);
    ws.h     = (float *)take((size_t)B * H * 4);
    ws.c     = (float *)take((size_t)B * H * 4);
    ws.h1    = (float *)take((size_t)B * H * 4);
    ws.c1    = (float *)take((size_t)B * H * 4);
    ws.gates = (float *)take((size_t)B * NG * 4);
    ws.A1    = (unsigned short *)take((size_t)B * KG * 2);
    ws.A2    = (unsigned short *)take((size_t)B * KG * 2);
    ws.Jb    = (unsigned short *)take((size_t)B * H * 2);
    ws.amax  = (unsigned long long *)take((size_t)B * 8);
    ws.label = (int *)take((size_t)B * 4);
    ws.bar   = (unsigned *)take(2 * 4);

    prep_kernel<<<2048, 256, 0, stream>>>(emb, Wih, Whh, bih, bhh, Wenc, benc,
                                          Wpred, bpred, Wout, bout, h0, c0, ws);
    decode_kernel<<<NBLK, NTHR, 0, stream>>>(enc, lens, out, ws);
}
